// CNN_RNN_45320494907625
// MI455X (gfx1250) — matrix operand order repaired and checked by reading, not compiled, not run on hardware
//
#include <hip/hip_runtime.h>
#include <hip/hip_bf16.h>
#include <math.h>

// ---------------------------------------------------------------------------
// CNN-RNN pipeline for MI455X (gfx1250, wave32, WMMA).
//   B=64, T=512, E=768, H=512, 3H=1536, NF=256, FS={3,4,5}
// All GEMMs use v_wmma_f32_16x16x32_f16 (f16 in, f32 accumulate).
// Weights are kept in their native row-major [N][K] layout so the WMMA
// B-fragment is two contiguous 8-half (16 B) loads per lane.
// The three sequential GRU scans run as persistent single-WGP kernels with
// h double-buffered in LDS (f16) and fp32 h held in registers per wave.
// ---------------------------------------------------------------------------

typedef __attribute__((ext_vector_type(16))) _Float16 v16h;
typedef __attribute__((ext_vector_type(8)))  float    v8f;

#define Bsz 64
#define Tt  512
#define Ee  768
#define Hh  512
#define G3H 1536
#define NFf 256

__device__ __forceinline__ float sigmf(float x) { return 1.0f / (1.0f + __expf(-x)); }

// A fragment: 16x32 (MxK) f16 tile from row-major A with leading dim lda.
// Lane l holds row (row0 + (l&15)); halves 0..7 -> K = k0 + hi*8 + i,
// halves 8..15 -> K = k0 + 16 + hi*8 + i  (hi = l>>4).
__device__ __forceinline__ v16h load_a_frag(const _Float16* A, int lda, int row0,
                                            int k0, int lane) {
  int lo = lane & 15, hi = lane >> 4;
  const _Float16* p = A + (long)(row0 + lo) * lda + k0 + hi * 8;
  v16h a;
#pragma unroll
  for (int i = 0; i < 8; ++i) { a[i] = p[i]; a[8 + i] = p[16 + i]; }
  return a;
}

// B fragment: 32x16 (KxN) tile where the weight matrix is stored row-major
// [N][K] (the natural PyTorch/JAX W layout, W.T applied by the GEMM).
// Lane l holds column (col0 + (l&15)); halves 0..7 -> K = k0 + hi*8 + i,
// halves 8..15 -> K = k0 + 16 + hi*8 + i (matches the A fragment), no transpose pass needed.
__device__ __forceinline__ v16h load_b_frag(const _Float16* W, int ldk, int k0,
                                            int col0, int lane) {
  int lo = lane & 15, hi = lane >> 4;
  const _Float16* p = W + (long)(col0 + lo) * ldk + k0 + hi * 8;
  v16h b;
#pragma unroll
  for (int i = 0; i < 8; ++i) { b[i] = p[i]; b[8 + i] = p[16 + i]; }
  return b;
}

// ---------------------------------------------------------------------------
__global__ void f32_to_f16(const float* __restrict__ src, _Float16* __restrict__ dst,
                           long n) {
  long i = (long)blockIdx.x * blockDim.x + threadIdx.x;
  if (i < n) dst[i] = (_Float16)src[i];
}

__global__ void zero_i32(int* __restrict__ p, long n) {
  long i = (long)blockIdx.x * blockDim.x + threadIdx.x;
  if (i < n) p[i] = 0;
}

// ---------------------------------------------------------------------------
// C_f16[M,N] = A_f16[M,K] @ W_f16[N,K]^T + bias_f32[N]
// 8 waves / block, one 16x16 output tile per wave. M%16==0, N%16==0, K%32==0,
// total tiles divisible by 8 (true for all uses here) -> EXEC all-ones always.
__global__ void wmma_gemm_bias_f16(const _Float16* __restrict__ A,
                                   const _Float16* __restrict__ W,
                                   const float* __restrict__ bias,
                                   _Float16* __restrict__ C,
                                   int M, int N, int K) {
  int lane = threadIdx.x & 31, wave = threadIdx.x >> 5;
  int lo = lane & 15, hi = lane >> 4;
  int tilesN = N >> 4;
  int tile = blockIdx.x * 8 + wave;
  int tm = (tile / tilesN) << 4;
  int tn = (tile % tilesN) << 4;
  v8f acc = {};
  for (int k0 = 0; k0 < K; k0 += 32) {
    __builtin_prefetch((const void*)(A + (long)(tm + lo) * K + k0 + 64), 0, 0);
    v16h a  = load_a_frag(A, K, tm, k0, lane);
    v16h bf = load_b_frag(W, K, k0, tn, lane);
    acc = __builtin_amdgcn_wmma_f32_16x16x32_f16(false, a, false, bf, (short)0, acc,
                                                 false, false);
  }
#pragma unroll
  for (int r = 0; r < 8; ++r) {
    int m = tm + r + 8 * hi, n = tn + lo;
    C[(long)m * N + n] = (_Float16)(acc[r] + bias[n]);
  }
}

// ---------------------------------------------------------------------------
// Persistent GRU scan (one workgroup, 32 waves, all 512 steps).
//  GI  : precomputed x@Wih^T + bih, f16 [B*T, 3H]
//  Whh : f16 [3H, H] row-major
//  h   : f16 double-buffered in LDS (A-operand for WMMA), fp32 copy of each
//        wave's own 4 tiles kept in registers across steps.
// Tasks: 4 M-tiles x 32 col-groups = 128 tile-tasks; wave w owns tasks
// w*4..w*4+3, each task computes the r/z/n gate tiles (3 WMMA chains, K=512)
// and fuses the GRU pointwise update for its 16 hidden columns.
__global__ void __launch_bounds__(1024) gru_scan(
    const _Float16* __restrict__ GI,
    const _Float16* __restrict__ Whh,
    const float* __restrict__ bhh,
    const int* __restrict__ nsel,     // layer mode: valid length per batch
    const float* __restrict__ Ws,     // selector mode: [2,H]
    const float* __restrict__ bs,     // selector mode: [2]
    int* __restrict__ sel_out,        // selector mode: [B,T]
    _Float16* __restrict__ ys_out,    // layer mode: [B,T,H] f16
    int selector_mode) {
  extern __shared__ _Float16 hbuf[];  // 2 * B*H halves (double buffer)
  const int BH = Bsz * Hh;
  int tid = threadIdx.x;
  int lane = tid & 31, wave = tid >> 5;
  int lo = lane & 15, hi = lane >> 4;

  for (int i = tid; i < 2 * BH; i += blockDim.x) hbuf[i] = (_Float16)0.0f;
  __syncthreads();

  float hreg[4][8];
#pragma unroll
  for (int i = 0; i < 4; ++i) {
#pragma unroll
    for (int r = 0; r < 8; ++r) hreg[i][r] = 0.0f;
  }

  for (int t = 0; t < Tt; ++t) {
    _Float16* cur = hbuf + (t & 1) * BH;
    _Float16* nxt = hbuf + ((t + 1) & 1) * BH;
#pragma unroll
    for (int i = 0; i < 4; ++i) {
      int task = wave * 4 + i;
      int mt = (task >> 5) << 4;  // batch-row tile base (0,16,32,48)
      int cg = (task & 31) << 4;  // hidden column base (0..496)
      v8f ar = {}; v8f az = {}; v8f an = {};
      for (int k0 = 0; k0 < Hh; k0 += 32) {
        v16h a  = load_a_frag(cur, Hh, mt, k0, lane);      // LDS reads
        v16h br = load_b_frag(Whh, Hh, k0, cg, lane);      // L2-resident weights
        ar = __builtin_amdgcn_wmma_f32_16x16x32_f16(false, a, false, br, (short)0,
                                                    ar, false, false);
        v16h bz = load_b_frag(Whh, Hh, k0, Hh + cg, lane);
        az = __builtin_amdgcn_wmma_f32_16x16x32_f16(false, a, false, bz, (short)0,
                                                    az, false, false);
        v16h bn = load_b_frag(Whh, Hh, k0, 2 * Hh + cg, lane);
        an = __builtin_amdgcn_wmma_f32_16x16x32_f16(false, a, false, bn, (short)0,
                                                    an, false, false);
      }
      int j = cg + lo;  // hidden unit owned by this lane
#pragma unroll
      for (int r = 0; r < 8; ++r) {
        int b = mt + r + 8 * hi;  // batch row
        long gbase = ((long)b * Tt + t) * (long)G3H;
        float gir = (float)GI[gbase + j];
        float giz = (float)GI[gbase + Hh + j];
        float gin = (float)GI[gbase + 2 * Hh + j];
        float ghr = ar[r] + bhh[j];
        float ghz = az[r] + bhh[Hh + j];
        float ghn = an[r] + bhh[2 * Hh + j];
        float rr = sigmf(gir + ghr);
        float zz = sigmf(giz + ghz);
        float nn = tanhf(gin + rr * ghn);
        float hn2 = (1.0f - zz) * nn + zz * hreg[i][r];
        if (!selector_mode) {
          int ok = t < nsel[b];
          ys_out[((long)b * Tt + t) * Hh + j] = (_Float16)(ok ? hn2 : 0.0f);
          hn2 = ok ? hn2 : hreg[i][r];
        }
        hreg[i][r] = hn2;
        nxt[b * Hh + j] = (_Float16)hn2;
      }
    }
    __syncthreads();
    if (selector_mode) {
      if (tid < Bsz) {
        const _Float16* hb = nxt + tid * Hh;
        float l0 = bs[0], l1 = bs[1];
        for (int k = 0; k < Hh; ++k) {
          float hv = (float)hb[k];
          l0 += hv * Ws[k];
          l1 += hv * Ws[Hh + k];
        }
        sel_out[tid * Tt + t] = (l1 > l0) ? 1 : 0;  // argmax, ties -> 0
      }
      __syncthreads();
    }
  }
}

// ---------------------------------------------------------------------------
// Selection forcing + stable compaction. One block per batch row.
__global__ void select_compact(const _Float16* __restrict__ emb,
                               const int* __restrict__ sel_in,
                               const int* __restrict__ mask,
                               _Float16* __restrict__ new_emb,
                               int* __restrict__ nsel_out) {
  __shared__ int s_sel[Tt];
  __shared__ int s_dst[Tt];
  __shared__ int s_len;
  __shared__ int s_n;
  int b = blockIdx.x, tid = threadIdx.x;
  if (tid == 0) {
    int len = 0;
    for (int t = 0; t < Tt; ++t) len += (mask[b * Tt + t] != 0);
    s_len = len;
  }
  __syncthreads();
  int len = s_len;
  for (int t = tid; t < Tt; t += blockDim.x) {
    int s = sel_in[b * Tt + t];
    if (t == 0) s = 1;
    if (t == len - 1) s = 1;
    if (t >= len) s = 0;
    s_sel[t] = s;
  }
  __syncthreads();
  if (tid == 0) {
    int acc = 0;
    for (int t = 0; t < Tt; ++t) { s_dst[t] = acc; acc += s_sel[t]; }
    s_n = acc;
    nsel_out[b] = acc;
  }
  __syncthreads();
  int n = s_n;
  for (long idx = tid; idx < (long)Tt * Ee; idx += blockDim.x) {
    int t = (int)(idx / Ee), e = (int)(idx % Ee);
    if (s_sel[t])
      new_emb[((long)b * Tt + s_dst[t]) * Ee + e] = emb[((long)b * Tt + t) * Ee + e];
  }
  for (long idx = tid; idx < (long)(Tt - n) * Ee; idx += blockDim.x) {
    int t = n + (int)(idx / Ee), e = (int)(idx % Ee);
    new_emb[((long)b * Tt + t) * Ee + e] = (_Float16)0.0f;
  }
}

// ---------------------------------------------------------------------------
// Conv1d(F taps over H channels) as sliding-window GEMM, fused bias+ReLU and
// time-max-pool via non-negative-float bit-pattern atomicMax.
// Window rows are contiguous F*H f16 runs in ys ([B*T, H]).
// Grid: B * 32 tTiles * 16 fTiles / 8 waves = 4096 blocks.
__global__ void conv_relu_maxpool(const _Float16* __restrict__ ys,
                                  const _Float16* __restrict__ Wc,  // [NF, F*H]
                                  const float* __restrict__ bc,     // [NF]
                                  int* __restrict__ pooled,         // [B,768] f32 bits
                                  int F, int fsOff) {
  int lane = threadIdx.x & 31, wave = threadIdx.x >> 5;
  int lo = lane & 15, hi = lane >> 4;
  int tile = blockIdx.x * 8 + wave;
  int b  = tile >> 9;
  int tt = (tile >> 4) & 31;
  int tf = tile & 15;
  int t0 = tt << 4;
  int K = F * Hh;
  int tmax = Tt - F;                    // last valid window start
  int trow = t0 + lo; if (trow > tmax) trow = tmax;  // clamp: safe dup, masked below
  const _Float16* Arow = ys + ((long)b * Tt + trow) * Hh;
  v8f acc = {};
  for (int k0 = 0; k0 < K; k0 += 32) {
    v16h a;
#pragma unroll
    for (int i = 0; i < 8; ++i) { a[i] = Arow[k0 + hi * 8 + i]; a[8 + i] = Arow[k0 + 16 + hi * 8 + i]; }
    v16h bf = load_b_frag(Wc, K, k0, tf << 4, lane);
    acc = __builtin_amdgcn_wmma_f32_16x16x32_f16(false, a, false, bf, (short)0, acc,
                                                 false, false);
  }
  int f = (tf << 4) + lo;
  float bia = bc[f];
  float mx = 0.0f;
#pragma unroll
  for (int r = 0; r < 8; ++r) {
    int t = t0 + r + 8 * hi;
    float v = acc[r] + bia;
    v = v > 0.0f ? v : 0.0f;            // ReLU (>=0, so bit-compare == fp-compare)
    if (t > tmax) v = 0.0f;             // mask clamped/OOB windows
    mx = v > mx ? v : mx;
  }
  atomicMax(&pooled[b * (3 * NFf) + fsOff + f], __float_as_int(mx));
}

// ---------------------------------------------------------------------------
__global__ void head_kernel(const int* __restrict__ pooled,
                            const float* __restrict__ Wo,
                            const float* __restrict__ bo,
                            float* __restrict__ out) {
  __shared__ float red[256];
  int b = blockIdx.x, tid = threadIdx.x;
  float s = 0.0f;
  for (int i = tid; i < 3 * NFf; i += 256)
    s += __int_as_float(pooled[b * (3 * NFf) + i]) * Wo[i];
  red[tid] = s;
  __syncthreads();
  for (int st = 128; st > 0; st >>= 1) {
    if (tid < st) red[tid] += red[tid + st];
    __syncthreads();
  }
  if (tid == 0) out[b] = red[0] + bo[0];
}

// ---------------------------------------------------------------------------
extern "C" void kernel_launch(void* const* d_in, const int* in_sizes, int n_in,
                              void* d_out, int out_size, void* d_ws, size_t ws_size,
                              hipStream_t stream) {
  (void)in_sizes; (void)n_in; (void)out_size; (void)ws_size;
  const float* embedded = (const float*)d_in[0];
  const int*   mask     = (const int*)  d_in[1];
  const float* Wih_c = (const float*)d_in[2];
  const float* Whh_c = (const float*)d_in[3];
  const float* bih_c = (const float*)d_in[4];
  const float* bhh_c = (const float*)d_in[5];
  const float* Ws    = (const float*)d_in[6];
  const float* bs    = (const float*)d_in[7];
  const float* Wih0  = (const float*)d_in[8];
  const float* Whh0  = (const float*)d_in[9];
  const float* bih0  = (const float*)d_in[10];
  const float* bhh0  = (const float*)d_in[11];
  const float* Wih1  = (const float*)d_in[12];
  const float* Whh1  = (const float*)d_in[13];
  const float* bih1  = (const float*)d_in[14];
  const float* bhh1  = (const float*)d_in[15];
  const float* Wc3   = (const float*)d_in[16];
  const float* bc3   = (const float*)d_in[17];
  const float* Wc4   = (const float*)d_in[18];
  const float* bc4   = (const float*)d_in[19];
  const float* Wc5   = (const float*)d_in[20];
  const float* bc5   = (const float*)d_in[21];
  const float* Wo    = (const float*)d_in[22];
  const float* bo    = (const float*)d_in[23];

  char* base = (char*)d_ws;
  size_t off = 0;
  auto take = [&](size_t bytes) -> void* {
    void* p = (void*)(base + off);
    off = (off + bytes + 255) & ~(size_t)255;
    return p;
  };

  const long nEmb = (long)Bsz * Tt * Ee;          // 25,165,824
  const long nBT  = (long)Bsz * Tt;               // 32,768
  _Float16* emb_h    = (_Float16*)take(nEmb * 2);
  _Float16* wih_c_h  = (_Float16*)take((size_t)G3H * Ee * 2);
  _Float16* whh_c_h  = (_Float16*)take((size_t)G3H * Hh * 2);
  _Float16* wih0_h   = (_Float16*)take((size_t)G3H * Ee * 2);
  _Float16* whh0_h   = (_Float16*)take((size_t)G3H * Hh * 2);
  _Float16* wih1_h   = (_Float16*)take((size_t)G3H * Hh * 2);
  _Float16* whh1_h   = (_Float16*)take((size_t)G3H * Hh * 2);
  _Float16* wc3_h    = (_Float16*)take((size_t)NFf * 3 * Hh * 2);
  _Float16* wc4_h    = (_Float16*)take((size_t)NFf * 4 * Hh * 2);
  _Float16* wc5_h    = (_Float16*)take((size_t)NFf * 5 * Hh * 2);
  _Float16* GI       = (_Float16*)take((size_t)nBT * G3H * 2);   // reused 3x
  int*      sel      = (int*)     take((size_t)nBT * 4);
  int*      nsel     = (int*)     take(256);
  _Float16* new_emb  = (_Float16*)take(nEmb * 2);
  _Float16* ys0      = (_Float16*)take((size_t)nBT * Hh * 2);
  _Float16* ys1      = (_Float16*)take((size_t)nBT * Hh * 2);
  int*      pooled   = (int*)     take((size_t)Bsz * 3 * NFf * 4);

  auto cvt = [&](const float* s, _Float16* d, long n) {
    f32_to_f16<<<(int)((n + 255) / 256), 256, 0, stream>>>(s, d, n);
  };
  cvt(embedded, emb_h, nEmb);
  cvt(Wih_c, wih_c_h, (long)G3H * Ee);
  cvt(Whh_c, whh_c_h, (long)G3H * Hh);
  cvt(Wih0,  wih0_h,  (long)G3H * Ee);
  cvt(Whh0,  whh0_h,  (long)G3H * Hh);
  cvt(Wih1,  wih1_h,  (long)G3H * Hh);
  cvt(Whh1,  whh1_h,  (long)G3H * Hh);
  cvt(Wc3,   wc3_h,   (long)NFf * 3 * Hh);
  cvt(Wc4,   wc4_h,   (long)NFf * 4 * Hh);
  cvt(Wc5,   wc5_h,   (long)NFf * 5 * Hh);

  const int gemmBlocksE = (int)(((long)(nBT / 16) * (G3H / 16)) / 8);  // 24576
  const size_t scanShm = (size_t)2 * Bsz * Hh * sizeof(_Float16);      // 128 KB

  // Stage 1: selector GRU
  wmma_gemm_bias_f16<<<gemmBlocksE, 256, 0, stream>>>(emb_h, wih_c_h, bih_c, GI,
                                                      (int)nBT, G3H, Ee);
  gru_scan<<<1, 1024, scanShm, stream>>>(GI, whh_c_h, bhh_c, nsel, Ws, bs, sel, ys0, 1);
  select_compact<<<Bsz, 256, 0, stream>>>(emb_h, sel, mask, new_emb, nsel);

  // Stage 2: two masked GRU layers
  wmma_gemm_bias_f16<<<gemmBlocksE, 256, 0, stream>>>(new_emb, wih0_h, bih0, GI,
                                                      (int)nBT, G3H, Ee);
  gru_scan<<<1, 1024, scanShm, stream>>>(GI, whh0_h, bhh0, nsel, Ws, bs, sel, ys0, 0);
  wmma_gemm_bias_f16<<<gemmBlocksE, 256, 0, stream>>>(ys0, wih1_h, bih1, GI,
                                                      (int)nBT, G3H, Hh);
  gru_scan<<<1, 1024, scanShm, stream>>>(GI, whh1_h, bhh1, nsel, Ws, bs, sel, ys1, 0);

  // Stage 3: conv + ReLU + max-pool + head
  zero_i32<<<(int)((Bsz * 3 * NFf + 255) / 256), 256, 0, stream>>>(pooled,
                                                                   (long)Bsz * 3 * NFf);
  conv_relu_maxpool<<<4096, 256, 0, stream>>>(ys1, wc3_h, bc3, pooled, 3, 0);
  conv_relu_maxpool<<<4096, 256, 0, stream>>>(ys1, wc4_h, bc4, pooled, 4, NFf);
  conv_relu_maxpool<<<4096, 256, 0, stream>>>(ys1, wc5_h, bc5, pooled, 5, 2 * NFf);
  head_kernel<<<Bsz, 256, 0, stream>>>(pooled, Wo, bo, (float*)d_out);
}
